// ShiftedWindowAttention_38749194944942
// MI455X (gfx1250) — matrix operand order repaired and checked by reading, not compiled, not run on hardware
//
#include <hip/hip_runtime.h>

// Shifted-window attention (Swin) for gfx1250 / MI455X.
// Pipeline: prep (weights->f16, bias gather) -> qkv (fused shift-gather +
// 3 projections, f16 WMMA) -> attn (S=qk^T, reg softmax, O=PV, out-proj, unshift).
// Workspace use: 1MB (weights f16 + bias) + 3 * 128MB (q, k, vT f16) ~= 404MB.

typedef __attribute__((ext_vector_type(16))) _Float16 v16h;
typedef __attribute__((ext_vector_type(8)))  float    v8f;

__device__ __forceinline__ v8f wmma16(v16h a, v16h b, v8f c) {
  return __builtin_amdgcn_wmma_f32_16x16x32_f16(false, a, false, b, (short)0, c,
                                                false, false);
}

// A fragment (16x32 f16, M x K), source row-major with given half-stride.
// Lane L: m = L&15, half = L>>4; vgpr j holds K-pair k = half*16 + j*2.
__device__ __forceinline__ v16h load_afrag(const _Float16* base, int stride, int lane) {
  const int m = lane & 15;
  const int half = lane >> 4;
  const _Float16* row = base + m * stride + half * 16;
  union { v16h v; unsigned int u[8]; } r;
#pragma unroll
  for (int j = 0; j < 8; ++j) {
    const int k = j << 1;
    r.u[j] = *(const unsigned int*)(row + k);
  }
  return r.v;
}

// B fragment (32x16 f16, K x N) from B^T stored row-major [n][k]:
// lane L reads 16 contiguous halves at row n = L&15, offset (L>>4)*16. 32B aligned.
__device__ __forceinline__ v16h load_bfrag(const _Float16* base, int stride, int lane) {
  const int n = lane & 15;
  const int half = lane >> 4;
  return *(const v16h*)(base + n * stride + half * 16);
}

// Shift-mask region id on the shifted 112x112 grid (boundaries 105=112-7, 109=112-3).
__device__ __forceinline__ int regid(int x) {
  return (x < 105) ? 0 : ((x < 109) ? 1 : 2);
}

// ---------------------------------------------------------------------------
// prep: weights f32 -> f16 (kept row-major == B^T layout), bias gather.
// ---------------------------------------------------------------------------
__global__ void prep_kernel(const float* __restrict__ Qw, const float* __restrict__ Kw,
                            const float* __restrict__ Vw, const float* __restrict__ Pw,
                            const float* __restrict__ rpb, const int* __restrict__ rpi,
                            _Float16* __restrict__ wq, _Float16* __restrict__ wk,
                            _Float16* __restrict__ wv, _Float16* __restrict__ wp,
                            float* __restrict__ biasTab)
{
  const int id = blockIdx.x * 256 + threadIdx.x;
  if (id < 65536) {
    wq[id] = (_Float16)Qw[id];
  } else if (id < 131072) {
    const int i = id - 65536;  wk[i] = (_Float16)Kw[i];
  } else if (id < 196608) {
    const int i = id - 131072; wv[i] = (_Float16)Vw[i];
  } else if (id < 262144) {
    const int i = id - 196608; wp[i] = (_Float16)Pw[i];
  } else if (id < 262144 + 8 * 49 * 49) {
    const int t = id - 262144;
    const int h = t / 2401, ij = t - h * 2401;
    biasTab[t] = rpb[rpi[ij] * 8 + h];  // [h][i][j]
  }
}

// ---------------------------------------------------------------------------
// qkv: one workgroup (8 waves) per window. Gathers the shifted window into
// LDS as f16 (row-major, stride 264 halves -> conflict-free A-frag loads),
// then GEMM M=64(pad49) N=256 K=256 per projection. Wave w: mtile=w&3,
// ntiles (w>>2)*8 .. +8. B frags stream from L2-resident f16 weights.
// q: [win][head][tok64][32]; k: same; v: transposed [win][head][32][tok64].
// ---------------------------------------------------------------------------
__global__ void __launch_bounds__(256)
qkv_kernel(const float* __restrict__ Q, const float* __restrict__ K,
           const float* __restrict__ V,
           const _Float16* __restrict__ wq, const _Float16* __restrict__ wk,
           const _Float16* __restrict__ wv,
           const float* __restrict__ qb, const float* __restrict__ kb,
           const float* __restrict__ vb,
           _Float16* __restrict__ qbuf, _Float16* __restrict__ kbuf,
           _Float16* __restrict__ vT)
{
  __shared__ _Float16 sX[64 * 264];
  const int win = blockIdx.x;
  const int wc = win & 15, wr = (win >> 4) & 15, b = win >> 8;
  const int tid = threadIdx.x;
  const int lane = tid & 31, wid = tid >> 5;
  const int mtile = wid & 3, ntb = (wid >> 2) * 8;
  const int lj = lane & 15, half = lane >> 4;

#pragma unroll
  for (int p = 0; p < 3; ++p) {
    const float* X = (p == 0) ? Q : ((p == 1) ? K : V);
    const _Float16* Wm = (p == 0) ? wq : ((p == 1) ? wk : wv);
    const float* bias = (p == 0) ? qb : ((p == 1) ? kb : vb);

    __syncthreads();
    // Stage window (roll(-3,-3) + partition), rows 49..63 zero-padded.
    for (int row = 0; row < 64; ++row) {
      float val = 0.0f;
      if (row < 49) {
        int hs = wr * 7 + row / 7 + 3; if (hs >= 112) hs -= 112;
        int wsp = wc * 7 + row % 7 + 3; if (wsp >= 112) wsp -= 112;
        val = X[(((size_t)b * 112 + hs) * 112 + wsp) * 256 + tid];
      }
      sX[row * 264 + tid] = (_Float16)val;
    }
    __syncthreads();

    v8f acc[8] = {};
#pragma unroll
    for (int ks = 0; ks < 8; ++ks) {
      v16h a = load_afrag(sX + mtile * 16 * 264 + ks * 32, 264, lane);
#pragma unroll
      for (int nt = 0; nt < 8; ++nt) {
        v16h bf = load_bfrag(Wm + (ntb + nt) * 16 * 256 + ks * 32, 256, lane);
        acc[nt] = wmma16(a, bf, acc[nt]);
      }
    }

    const float scale = (p == 0) ? 0.17677669529663687f : 1.0f;  // hd^-0.5
#pragma unroll
    for (int nt = 0; nt < 8; ++nt) {
      const int c = (ntb + nt) * 16 + lj;
      const float bv = bias[c];
      const int head = c >> 5, hd = c & 31;
#pragma unroll
      for (int r = 0; r < 8; ++r) {
        const int tok = mtile * 16 + half * 8 + r;
        const _Float16 hv = (_Float16)((acc[nt][r] + bv) * scale);
        if (p == 2)
          vT[(((size_t)win * 8 + head) * 32 + hd) * 64 + tok] = hv;      // [hd][tok]
        else {
          _Float16* dst = (p == 0) ? qbuf : kbuf;
          dst[(((size_t)win * 8 + head) * 64 + tok) * 32 + hd] = hv;     // [tok][hd]
        }
      }
    }
  }
}

// ---------------------------------------------------------------------------
// attn: one workgroup per window; wave h == head h.
//   S = q k^T (16 wmma) -> +bias +analytic shift-mask -> register softmax
//   (shfl_xor across 16-lane groups) -> P to LDS (C->A transpose) ->
//   O = P V (16 wmma) -> LDS -> fused output projection (64 wmma) ->
//   un-shift scatter to d_out.
// ---------------------------------------------------------------------------
__global__ void __launch_bounds__(256)
attn_kernel(const _Float16* __restrict__ qbuf, const _Float16* __restrict__ kbuf,
            const _Float16* __restrict__ vT, const float* __restrict__ biasTab,
            const _Float16* __restrict__ wp, const float* __restrict__ pbias,
            float* __restrict__ out)
{
  __shared__ union {
    _Float16 P[8][64 * 66];   // per-head attention probs, stride 66
    _Float16 O[64 * 264];     // projection A staging, stride 264
  } sm;

  const int win = blockIdx.x;
  const int wc = win & 15, wr = (win >> 4) & 15, b = win >> 8;
  const int tid = threadIdx.x;
  const int lane = tid & 31, h = tid >> 5;
  const int lj = lane & 15, half = lane >> 4;

  const _Float16* qh = qbuf + (((size_t)win * 8 + h) << 11);  // 64*32
  const _Float16* kh = kbuf + (((size_t)win * 8 + h) << 11);
  const _Float16* vh = vT + (((size_t)win * 8 + h) << 11);    // 32*64

  // ---- S = q @ k^T (64x64 in 16 f32 tiles) ----
  v8f s[4][4] = {};
#pragma unroll
  for (int mt = 0; mt < 4; ++mt) {
    v16h a = load_afrag(qh + mt * 16 * 32, 32, lane);
#pragma unroll
    for (int nt = 0; nt < 4; ++nt) {
      v16h bf = load_bfrag(kh + nt * 16 * 32, 32, lane);
      s[mt][nt] = wmma16(a, bf, s[mt][nt]);
    }
  }

  // ---- relative-position bias + shift mask (computed analytically) ----
#pragma unroll
  for (int nt = 0; nt < 4; ++nt) {
    const int jj = nt * 16 + lj;
    const bool jok = jj < 49;
    int idj = 0;
    if (jok) idj = regid(wr * 7 + jj / 7) * 3 + regid(wc * 7 + jj % 7);
#pragma unroll
    for (int mt = 0; mt < 4; ++mt) {
#pragma unroll
      for (int r = 0; r < 8; ++r) {
        const int ii = mt * 16 + half * 8 + r;
        float v = s[mt][nt][r];
        if (!jok) v = -1e30f;  // pad keys -> softmax 0
        else if (ii < 49) {
          v += biasTab[(h * 49 + ii) * 49 + jj];
          const int idi = regid(wr * 7 + ii / 7) * 3 + regid(wc * 7 + ii % 7);
          if (idi != idj) v -= 100.0f;
        }
        s[mt][nt][r] = v;
      }
    }
  }

  // ---- softmax over j: row lives in 4 tiles x 16 lanes (fixed half) ----
#pragma unroll
  for (int mt = 0; mt < 4; ++mt) {
#pragma unroll
    for (int r = 0; r < 8; ++r) {
      float a0 = s[mt][0][r], a1 = s[mt][1][r], a2 = s[mt][2][r], a3 = s[mt][3][r];
      float mx = fmaxf(fmaxf(a0, a1), fmaxf(a2, a3));
#pragma unroll
      for (int off = 1; off < 16; off <<= 1)
        mx = fmaxf(mx, __shfl_xor(mx, off, 32));
      const float e0 = __expf(a0 - mx), e1 = __expf(a1 - mx);
      const float e2 = __expf(a2 - mx), e3 = __expf(a3 - mx);
      float sum = e0 + e1 + e2 + e3;
#pragma unroll
      for (int off = 1; off < 16; off <<= 1)
        sum += __shfl_xor(sum, off, 32);
      const float inv = 1.0f / sum;
      s[mt][0][r] = e0 * inv; s[mt][1][r] = e1 * inv;
      s[mt][2][r] = e2 * inv; s[mt][3][r] = e3 * inv;
    }
  }

  // ---- P -> LDS (per-head, row-major [i][j], stride 66) ----
  _Float16* pw = sm.P[h];
#pragma unroll
  for (int mt = 0; mt < 4; ++mt)
#pragma unroll
    for (int nt = 0; nt < 4; ++nt)
#pragma unroll
      for (int r = 0; r < 8; ++r)
        pw[(mt * 16 + half * 8 + r) * 66 + nt * 16 + lj] = (_Float16)s[mt][nt][r];

  // ---- O = P @ V (64x32) ----
  v8f o[4][2] = {};
#pragma unroll
  for (int mt = 0; mt < 4; ++mt) {
#pragma unroll
    for (int ks = 0; ks < 2; ++ks) {
      v16h a = load_afrag(pw + mt * 16 * 66 + ks * 32, 66, lane);
#pragma unroll
      for (int nt = 0; nt < 2; ++nt) {
        v16h bf = load_bfrag(vh + nt * 16 * 64 + ks * 32, 64, lane);
        o[mt][nt] = wmma16(a, bf, o[mt][nt]);
      }
    }
  }

  __syncthreads();  // all waves done with sm.P before it aliases sm.O

  // ---- O -> LDS row-major [tok][channel], stride 264 ----
#pragma unroll
  for (int mt = 0; mt < 4; ++mt)
#pragma unroll
    for (int nt = 0; nt < 2; ++nt)
#pragma unroll
      for (int r = 0; r < 8; ++r)
        sm.O[(mt * 16 + half * 8 + r) * 264 + h * 32 + nt * 16 + lj] =
            (_Float16)o[mt][nt][r];

  __syncthreads();

  // ---- fused output projection: (64x256) @ Pw^T, + Pb, un-shift scatter ----
  const int mtile = h & 3, ntb = (h >> 2) * 8;
  v8f acc[8] = {};
#pragma unroll
  for (int ks = 0; ks < 8; ++ks) {
    v16h a = load_afrag(sm.O + mtile * 16 * 264 + ks * 32, 264, lane);
#pragma unroll
    for (int nt = 0; nt < 8; ++nt) {
      v16h bf = load_bfrag(wp + (ntb + nt) * 16 * 256 + ks * 32, 256, lane);
      acc[nt] = wmma16(a, bf, acc[nt]);
    }
  }
#pragma unroll
  for (int nt = 0; nt < 8; ++nt) {
    const int c = (ntb + nt) * 16 + lj;
    const float bv = pbias[c];
#pragma unroll
    for (int r = 0; r < 8; ++r) {
      const int tok = mtile * 16 + half * 8 + r;
      if (tok < 49) {
        int oh = wr * 7 + tok / 7 + 3; if (oh >= 112) oh -= 112;   // roll(+3)
        int ow = wc * 7 + tok % 7 + 3; if (ow >= 112) ow -= 112;
        out[(((size_t)b * 112 + oh) * 112 + ow) * 256 + c] = acc[nt][r] + bv;
      }
    }
  }
}

// ---------------------------------------------------------------------------
extern "C" void kernel_launch(void* const* d_in, const int* in_sizes, int n_in,
                              void* d_out, int out_size, void* d_ws, size_t ws_size,
                              hipStream_t stream) {
  const float* Q   = (const float*)d_in[0];
  const float* K   = (const float*)d_in[1];
  const float* V   = (const float*)d_in[2];
  const float* Qw  = (const float*)d_in[3];
  const float* Qb  = (const float*)d_in[4];
  const float* Kw  = (const float*)d_in[5];
  const float* Kb  = (const float*)d_in[6];
  const float* Vw  = (const float*)d_in[7];
  const float* Vb  = (const float*)d_in[8];
  const float* Pw  = (const float*)d_in[9];
  const float* Pb  = (const float*)d_in[10];
  const float* rpb = (const float*)d_in[11];
  const int*   rpi = (const int*)d_in[12];
  float* out = (float*)d_out;

  char* ws = (char*)d_ws;
  _Float16* wq = (_Float16*)(ws + ((size_t)0 << 17));
  _Float16* wk = (_Float16*)(ws + ((size_t)1 << 17));
  _Float16* wv = (_Float16*)(ws + ((size_t)2 << 17));
  _Float16* wp = (_Float16*)(ws + ((size_t)3 << 17));
  float* biasTab = (float*)(ws + ((size_t)4 << 17));   // 8*49*49 f32
  const size_t BUF = (size_t)4096 * 8 * 64 * 32;       // halves per q/k/v buffer
  _Float16* qbuf = (_Float16*)(ws + ((size_t)1 << 20));
  _Float16* kbuf = qbuf + BUF;
  _Float16* vTb  = kbuf + BUF;

  prep_kernel<<<1100, 256, 0, stream>>>(Qw, Kw, Vw, Pw, rpb, rpi,
                                        wq, wk, wv, wp, biasTab);
  qkv_kernel<<<4096, 256, 0, stream>>>(Q, K, V, wq, wk, wv, Qb, Kb, Vb,
                                       qbuf, kbuf, vTb);
  attn_kernel<<<4096, 256, 0, stream>>>(qbuf, kbuf, vTb, biasTab, wp, Pb, out);
}
